// QuantumQNetwork_74182675137099
// MI455X (gfx1250) — matrix operand order repaired and checked by reading, compiled, not run on hardware
//
#include <hip/hip_runtime.h>
#include <cstdint>

#define WAVES_PER_BLOCK 8
#define ROWS_PER_WAVE   32
#define ROWS_PER_BLOCK  256

typedef __attribute__((ext_vector_type(4)))  float    v4f;
typedef __attribute__((ext_vector_type(8)))  float    v8f;
typedef __attribute__((ext_vector_type(16))) _Float16 v16h;
typedef __attribute__((ext_vector_type(8)))  _Float16 v8h;
typedef __attribute__((ext_vector_type(4)))  _Float16 v4h;

static __device__ __forceinline__ v16h cat16(v8h a, v8h b) {
  return __builtin_shufflevector(a, b, 0,1,2,3,4,5,6,7,8,9,10,11,12,13,14,15);
}
static __device__ __forceinline__ v8h pack8(v4f a, v4f b) {
  v8h o;
  o[0]=(_Float16)a[0]; o[1]=(_Float16)a[1]; o[2]=(_Float16)a[2]; o[3]=(_Float16)a[3];
  o[4]=(_Float16)b[0]; o[5]=(_Float16)b[1]; o[6]=(_Float16)b[2]; o[7]=(_Float16)b[3];
  return o;
}
static __device__ __forceinline__ v4h pack4(v4f a) {
  v4h o; o[0]=(_Float16)a[0]; o[1]=(_Float16)a[1]; o[2]=(_Float16)a[2]; o[3]=(_Float16)a[3];
  return o;
}
static __device__ __forceinline__ float tanh_fast(float x) {
  float e = __expf(2.0f * x);
  return __fdividef(e - 1.0f, e + 1.0f);
}
// CDNA5 async global->LDS copy (ASYNCcnt path)
static __device__ __forceinline__ void async_g2l_b32(void* lds, const void* g) {
  unsigned int l = (unsigned int)(uintptr_t)lds;           // low 32 bits = LDS byte addr
  unsigned long long ga = (unsigned long long)(uintptr_t)g;
  asm volatile("global_load_async_to_lds_b32 %0, %1, off" :: "v"(l), "v"(ga) : "memory");
}
static __device__ __forceinline__ void wait_async0() {
  asm volatile("s_wait_asynccnt 0" ::: "memory");
}

// RY on wire with pair mask (8>>wire). Fully unrolled, mask is compile-time.
static __device__ __forceinline__ void apply_ry(float (&st)[16], float th, int mask) {
  float sn, cs;
  __sincosf(0.5f * th, &sn, &cs);
#pragma unroll
  for (int i = 0; i < 16; ++i) {
    if (!(i & mask)) {
      float a = st[i], b = st[i | mask];
      st[i]        = cs * a - sn * b;
      st[i | mask] = sn * a + cs * b;
    }
  }
}
static __device__ __forceinline__ void cnot(float (&st)[16], int cm, int tm) {
#pragma unroll
  for (int i = 0; i < 16; ++i) {
    if ((i & cm) && !(i & tm)) {
      float tmp = st[i]; st[i] = st[i | tm]; st[i | tm] = tmp;
    }
  }
}

__global__ __launch_bounds__(256) void qnet_fused(
    const float* __restrict__ x,  const float* __restrict__ gW1,
    const float* __restrict__ gb1, const float* __restrict__ gW2,
    const float* __restrict__ gb2, const float* __restrict__ gqp,
    const float* __restrict__ gH1, const float* __restrict__ gc1,
    const float* __restrict__ gH2, const float* __restrict__ gc2,
    float* __restrict__ out)
{
  __shared__ __align__(16) _Float16 sW1[32 * 64];   // W1 as f16 (WMMA B)
  __shared__ __align__(16) _Float16 sH2[16 * 32];   // H2 as f16 (WMMA B)
  __shared__ __align__(16) float sW2[128];
  __shared__ __align__(16) float sH1[128];
  __shared__ __align__(16) float sB1[32];
  __shared__ __align__(16) float sC1[32];
  __shared__ __align__(16) float sC2[16];
  __shared__ __align__(16) float sB2[4];
  __shared__ __align__(16) float sQP[8];
  __shared__ __align__(16) unsigned char sStage[WAVES_PER_BLOCK][4096]; // x(f16) then h(f32)
  __shared__ __align__(16) _Float16 sAct[WAVES_PER_BLOCK][32 * 32];     // h2 (f16)

  const int t    = threadIdx.x;
  const int wave = t >> 5;
  const int lane = t & 31;
  const int hi   = lane >> 4;   // half-wave select
  const int lo   = lane & 15;

  // ---- Phase 0: stage weights ----
  {
    int i = t * 8;                                   // 256*8 = 2048 = |W1|
    v4f a = *(const v4f*)(gW1 + i);
    v4f b = *(const v4f*)(gW1 + i + 4);
    *(v8h*)(sW1 + i) = pack8(a, b);
    if (t < 128) {
      int j = t * 4;                                 // 128*4 = 512 = |H2|
      v4f hc = *(const v4f*)(gH2 + j);
      *(v4h*)(sH2 + j) = pack4(hc);
    }
    // small f32 params via async global->LDS
    if (t < 128) async_g2l_b32(&sW2[t], gW2 + t);
    else         async_g2l_b32(&sH1[t - 128], gH1 + (t - 128));
    if (t < 32)      async_g2l_b32(&sB1[t],      gb1 + t);
    else if (t < 36) async_g2l_b32(&sB2[t - 32], gb2 + (t - 32));
    else if (t < 44) async_g2l_b32(&sQP[t - 36], gqp + (t - 36));
    else if (t < 76) async_g2l_b32(&sC1[t - 44], gc1 + (t - 44));
    else if (t < 92) async_g2l_b32(&sC2[t - 76], gc2 + (t - 76));
    wait_async0();
  }
  __syncthreads();

  // ---- constant B fragments (live whole kernel) ----
  v16h bW1[2][2];   // [kblock][ntile], B[k][n] = W1[n][k]
#pragma unroll
  for (int kb = 0; kb < 2; ++kb)
#pragma unroll
    for (int nt = 0; nt < 2; ++nt) {
      int n = nt * 16 + lo;
      const _Float16* p = sW1 + n * 64 + kb * 32 + hi * 8;
      bW1[kb][nt] = cat16(*(const v8h*)p, *(const v8h*)(p + 16));
    }
  v16h bH2;         // B[k][n] = H2[n][k]
  {
    const _Float16* p = sH2 + lo * 32 + hi * 8;
    bH2 = cat16(*(const v8h*)p, *(const v8h*)(p + 16));
  }

  // ---- Phase 1: load x tile (32 rows x 64 f32, contiguous) -> f16 LDS ----
  const int rowbase = blockIdx.x * ROWS_PER_BLOCK + wave * ROWS_PER_WAVE;
  _Float16* xh = (_Float16*)sStage[wave];
  {
    const float* xt = x + (size_t)rowbase * 64;
#pragma unroll
    for (int i = 0; i < 16; ++i) {
      int d = i * 128 + lane * 4;
      v4f f = *(const v4f*)(xt + d);
      *(v4h*)(xh + d) = pack4(f);
    }
  }
  __syncthreads();

  // ---- GEMM1: h = relu(x @ W1^T + b1), 8 WMMAs per wave ----
  v8f cacc[2][2];
#pragma unroll
  for (int tr = 0; tr < 2; ++tr) {
    v16h a[2];
#pragma unroll
    for (int kb = 0; kb < 2; ++kb) {
      int row = tr * 16 + lo;
      const _Float16* p = xh + row * 64 + kb * 32 + hi * 8;
      a[kb] = cat16(*(const v8h*)p, *(const v8h*)(p + 16));
    }
#pragma unroll
    for (int nt = 0; nt < 2; ++nt) {
      v8f c = {};
      c = __builtin_amdgcn_wmma_f32_16x16x32_f16(false, a[0], false, bW1[0][nt], (short)0, c, false, false);
      c = __builtin_amdgcn_wmma_f32_16x16x32_f16(false, a[1], false, bW1[1][nt], (short)0, c, false, false);
      cacc[tr][nt] = c;
    }
  }
  asm volatile("" ::: "memory");   // keep h-stores after xh-loads (buffer reuse)
  float* hA = (float*)sStage[wave];
  {
    float bcol[2] = { sB1[lo], sB1[16 + lo] };
#pragma unroll
    for (int tr = 0; tr < 2; ++tr)
#pragma unroll
      for (int nt = 0; nt < 2; ++nt)
#pragma unroll
        for (int r = 0; r < 8; ++r) {
          int rl  = tr * 16 + 8 * hi + r;
          int col = nt * 16 + lo;
          hA[rl * 32 + col] = fmaxf(cacc[tr][nt][r] + bcol[nt], 0.0f);
        }
  }
  __syncthreads();

  // ---- Phase 2 (per-lane, one row each): enc -> circuit -> zs -> h2 ----
  {
    float hrow[32];
#pragma unroll
    for (int i = 0; i < 8; ++i) {
      v4f v = *(const v4f*)(hA + lane * 32 + i * 4);
      hrow[i*4+0]=v[0]; hrow[i*4+1]=v[1]; hrow[i*4+2]=v[2]; hrow[i*4+3]=v[3];
    }
    float ea[4] = { sB2[0], sB2[1], sB2[2], sB2[3] };
#pragma unroll
    for (int k = 0; k < 32; ++k) {
      float hk = hrow[k];
#pragma unroll
      for (int w = 0; w < 4; ++w) ea[w] = fmaf(hk, sW2[w * 32 + k], ea[w]);
    }
    // statevector |0000>, RY encodings (wire w -> pair mask 8>>w)
    float st[16];
#pragma unroll
    for (int i = 0; i < 16; ++i) st[i] = 0.0f;
    st[0] = 1.0f;
#pragma unroll
    for (int w = 0; w < 4; ++w) apply_ry(st, tanh_fast(ea[w]), 8 >> w);
    float qp0[8];
#pragma unroll
    for (int i = 0; i < 8; ++i) qp0[i] = sQP[i];
#pragma unroll
    for (int layer = 0; layer < 2; ++layer) {
      cnot(st, 8, 4); cnot(st, 4, 2); cnot(st, 2, 1); cnot(st, 1, 8);
#pragma unroll
      for (int w = 0; w < 4; ++w) apply_ry(st, qp0[layer * 4 + w], 8 >> w);
    }
    float p[16];
#pragma unroll
    for (int i = 0; i < 16; ++i) p[i] = st[i] * st[i];
    float zs[4];
#pragma unroll
    for (int w = 0; w < 4; ++w) {
      int m = 8 >> w; float acc = 0.0f;
#pragma unroll
      for (int i = 0; i < 16; ++i) acc += (i & m) ? -p[i] : p[i];
      zs[w] = acc;
    }
    // h2 = relu(H1 @ zs + c1) -> f16 LDS
    _Float16 h2loc[32];
#pragma unroll
    for (int j = 0; j < 32; ++j) {
      float acc = sC1[j];
#pragma unroll
      for (int w = 0; w < 4; ++w) acc = fmaf(sH1[j * 4 + w], zs[w], acc);
      h2loc[j] = (_Float16)fmaxf(acc, 0.0f);
    }
    _Float16* dst = sAct[wave] + lane * 32;
#pragma unroll
    for (int i = 0; i < 4; ++i) {
      v8h o;
#pragma unroll
      for (int j = 0; j < 8; ++j) o[j] = h2loc[i * 8 + j];
      *(v8h*)(dst + i * 8) = o;
    }
  }
  __syncthreads();

  // ---- GEMM4: q = tanh(h2 @ H2^T + c2), 2 WMMAs per wave ----
  {
    float c2l = sC2[lo];
#pragma unroll
    for (int tr = 0; tr < 2; ++tr) {
      int row = tr * 16 + lo;
      const _Float16* p = sAct[wave] + row * 32 + hi * 8;
      v16h a = cat16(*(const v8h*)p, *(const v8h*)(p + 16));
      v8f c = {};
      c = __builtin_amdgcn_wmma_f32_16x16x32_f16(false, a, false, bH2, (short)0, c, false, false);
      int rg0 = rowbase + tr * 16 + 8 * hi;
#pragma unroll
      for (int r = 0; r < 8; ++r)
        out[(size_t)(rg0 + r) * 16 + lo] = tanh_fast(c[r] + c2l);
    }
  }
}

extern "C" void kernel_launch(void* const* d_in, const int* in_sizes, int n_in,
                              void* d_out, int out_size, void* d_ws, size_t ws_size,
                              hipStream_t stream) {
  const float* x  = (const float*)d_in[0];
  const float* W1 = (const float*)d_in[1];
  const float* b1 = (const float*)d_in[2];
  const float* W2 = (const float*)d_in[3];
  const float* b2 = (const float*)d_in[4];
  const float* qp = (const float*)d_in[5];
  const float* H1 = (const float*)d_in[6];
  const float* c1 = (const float*)d_in[7];
  const float* H2 = (const float*)d_in[8];
  const float* c2 = (const float*)d_in[9];
  float* out = (float*)d_out;

  int B = in_sizes[0] / 64;                 // 262144
  dim3 grid(B / ROWS_PER_BLOCK);            // 1024 workgroups
  qnet_fused<<<grid, 256, 0, stream>>>(x, W1, b1, W2, b2, qp, H1, c1, H2, c2, out);
}
